// PlacementGNN_73143293051349
// MI455X (gfx1250) — matrix operand order repaired and checked by reading, not compiled, not run on hardware
//
#include <hip/hip_runtime.h>
#include <hip/hip_bf16.h>

typedef __attribute__((ext_vector_type(16))) _Float16 v16h;
typedef __attribute__((ext_vector_type(8)))  _Float16 v8h;
typedef __attribute__((ext_vector_type(4)))  _Float16 v4h;
typedef __attribute__((ext_vector_type(8)))  float    v8f;
typedef __attribute__((ext_vector_type(4)))  float    f4;

#define HIDC 128
#define NHEAD 8
#define ENC_NEG_MAX 0x00800000u  /* fenc(-FLT_MAX) */

__device__ __forceinline__ unsigned fenc(float f){
  unsigned b = __float_as_uint(f);
  return (b & 0x80000000u) ? ~b : (b | 0x80000000u);
}
__device__ __forceinline__ float fdec(unsigned u){
  unsigned b = (u & 0x80000000u) ? (u & 0x7fffffffu) : ~u;
  return __uint_as_float(b);
}

// ---------------- encoders: out[m,c] = sum_k X[m,k]*W[k,c] + b[c] ----------------
__global__ void k_encode(const float* __restrict__ X, const float* __restrict__ W,
                         const float* __restrict__ b, float* __restrict__ out,
                         int M, int Kin){
  int m = blockIdx.x, c = threadIdx.x;
  if (m >= M) return;
  float s = b[c];
  for (int k = 0; k < Kin; ++k) s += X[(size_t)m*Kin + k] * W[k*HIDC + c];
  out[(size_t)m*HIDC + c] = s;
}

// ---------------- self-loop attr: mean of incoming encoded edge_attr ----------------
__global__ void k_loop_accum(const float* __restrict__ ea, const int* __restrict__ dst,
                             float* __restrict__ loop_attr, float* __restrict__ deg, int E){
  int e = blockIdx.x, c = threadIdx.x;
  if (e >= E) return;
  int d = dst[e];
  atomicAdd(&loop_attr[(size_t)d*HIDC + c], ea[(size_t)e*HIDC + c]);
  if (c == 0) atomicAdd(&deg[d], 1.0f);
}
__global__ void k_loop_div(float* __restrict__ loop_attr, const float* __restrict__ deg, int N){
  int n = blockIdx.x, c = threadIdx.x;
  if (n >= N) return;
  loop_attr[(size_t)n*HIDC + c] /= fmaxf(deg[n], 1.0f);
}

// ---------------- precontract attention vectors: v[k,h] = sum_c W[k,h*128+c]*att[h,c] ----------------
__global__ void k_attvec(const float* __restrict__ W, const float* __restrict__ att,
                         float* __restrict__ v){
  int k = threadIdx.x, h = blockIdx.x;   // 128 x 8
  float s = 0.f;
  for (int c = 0; c < HIDC; ++c) s += W[(size_t)k*1024 + h*HIDC + c] * att[h*HIDC + c];
  v[k*NHEAD + h] = s;
}

// -------- pack GAT head-merged weight into WMMA B-fragment order --------
// GEMM K index j = h*128+k, N index c. Fragment coords:
//   kstep s=j>>5, krel=j&31, tile t=c>>4, n=c&15, lane=n+16*(krel>>4), half=krel&15
__global__ void k_bcat_pack(const float* __restrict__ Wl, _Float16* __restrict__ Bp){
  int j = blockIdx.x, c = threadIdx.x;   // 1024 x 128
  int h = j >> 7, k = j & 127;
  float v = Wl[(size_t)k*1024 + h*HIDC + c];
  int s = j >> 5, krel = j & 31;
  int t = c >> 4, n = c & 15;
  int lane = n + ((krel >> 4) << 4);
  int half = krel & 15;
  Bp[((size_t)(s*8 + t)*32 + lane)*16 + half] = (_Float16)v;
}

// -------- pack a plain [K,Ncols] weight into WMMA B-fragment order --------
__global__ void k_pack_w(const float* __restrict__ W, _Float16* __restrict__ Bp, int Ncols){
  int k = blockIdx.x, c = threadIdx.x;   // K x Ncols
  float v = W[(size_t)k*Ncols + c];
  int s = k >> 5, krel = k & 31;
  int t = c >> 4, n = c & 15;
  int lane = n + ((krel >> 4) << 4);
  int half = krel & 15;
  Bp[((size_t)(s*(Ncols >> 4) + t)*32 + lane)*16 + half] = (_Float16)v;
}

// ---------------- per-node logits a_src/a_dst ----------------
__global__ void k_node_logits(const float* __restrict__ h, const float* __restrict__ vs,
                              const float* __restrict__ vd, float* __restrict__ asrc,
                              float* __restrict__ adst, int N){
  int idx = blockIdx.x*blockDim.x + threadIdx.x;
  if (idx >= N*NHEAD) return;
  int n = idx >> 3, hh = idx & 7;
  float s = 0.f, d = 0.f;
  for (int k = 0; k < HIDC; ++k){
    float x = h[(size_t)n*HIDC + k];
    s += x * vs[k*NHEAD + hh];
    d += x * vd[k*NHEAD + hh];
  }
  asrc[idx] = s; adst[idx] = d;
}

// ---------------- per-edge logit a_ed (edges + self loops) ----------------
__global__ void k_edge_logits(const float* __restrict__ ea, const float* __restrict__ loop_attr,
                              const float* __restrict__ ve, float* __restrict__ aed,
                              int E, int N){
  int idx = blockIdx.x*blockDim.x + threadIdx.x;
  if (idx >= (E + N)*NHEAD) return;
  int i = idx >> 3, hh = idx & 7;
  const float* row = (i < E) ? &ea[(size_t)i*HIDC] : &loop_attr[(size_t)(i - E)*HIDC];
  float s = 0.f;
  for (int k = 0; k < HIDC; ++k) s += row[k] * ve[k*NHEAD + hh];
  aed[idx] = s;
}

// ---------------- segment softmax ----------------
__global__ void k_smax_init(unsigned* __restrict__ maxb, float* __restrict__ sumb, int n){
  int i = blockIdx.x*blockDim.x + threadIdx.x;
  if (i < n){ maxb[i] = ENC_NEG_MAX; sumb[i] = 0.f; }
}
__global__ void k_logit_max(const int* __restrict__ src, const int* __restrict__ dst,
                            const float* __restrict__ asrc, const float* __restrict__ adst,
                            float* __restrict__ aed, unsigned* __restrict__ maxb, int E, int N){
  int idx = blockIdx.x*blockDim.x + threadIdx.x;
  if (idx >= (E + N)*NHEAD) return;
  int i = idx >> 3, hh = idx & 7;
  int s = (i < E) ? src[i] : (i - E);
  int d = (i < E) ? dst[i] : (i - E);
  float v = asrc[s*NHEAD + hh] + adst[d*NHEAD + hh] + aed[idx];
  v = (v > 0.f) ? v : 0.2f * v;   // leaky relu
  aed[idx] = v;
  atomicMax(&maxb[d*NHEAD + hh], fenc(v));
}
__global__ void k_exp_sum(const int* __restrict__ dst, const float* __restrict__ aed,
                          const unsigned* __restrict__ maxb, float* __restrict__ exb,
                          float* __restrict__ sumb, int E, int N){
  int idx = blockIdx.x*blockDim.x + threadIdx.x;
  if (idx >= (E + N)*NHEAD) return;
  int i = idx >> 3, hh = idx & 7;
  int d = (i < E) ? dst[i] : (i - E);
  float ex = __expf(aed[idx] - fdec(maxb[d*NHEAD + hh]));
  exb[idx] = ex;
  atomicAdd(&sumb[d*NHEAD + hh], ex);
}

// ---------------- aggregate raw features: z[d,h,k] += alpha * h[src,k] ----------------
__global__ void k_aggregate(const int* __restrict__ src, const int* __restrict__ dst,
                            const float* __restrict__ hprev, const float* __restrict__ exb,
                            const float* __restrict__ sumb, float* __restrict__ z, int E, int N){
  int i = blockIdx.x, k = threadIdx.x;
  int s = (i < E) ? src[i] : (i - E);
  int d = (i < E) ? dst[i] : (i - E);
  float val = hprev[(size_t)s*HIDC + k];
  #pragma unroll
  for (int hh = 0; hh < NHEAD; ++hh){
    float w = exb[i*NHEAD + hh] / (sumb[d*NHEAD + hh] + 1e-16f);
    atomicAdd(&z[(size_t)d*1024 + hh*HIDC + k], w * val);
  }
}

// --- shared helpers for WMMA GEMMs ---
__device__ __forceinline__ void stage_A_tile(_Float16 (*As)[40], const float* __restrict__ A,
                                             int m0, int M, int K, int kk, int tid){
  if (m0 + 64 <= M){                           // uniform fast path: no per-lane guards
    #pragma unroll
    for (int t = 0; t < 4; ++t){
      int e4 = tid + t*128;
      int r = e4 >> 3, c = (e4 & 7) * 4;
      const float* zp = &A[(size_t)(m0 + r)*K + kk + c];
      f4 v = *(const f4*)zp;
      if (kk + 32 < K) __builtin_prefetch(zp + 32, 0, 1);   // next K-slice
      v4h hv = { (_Float16)v.x, (_Float16)v.y, (_Float16)v.z, (_Float16)v.w };
      *(v4h*)&As[r][c] = hv;
    }
  } else {
    #pragma unroll
    for (int t = 0; t < 4; ++t){
      int e4 = tid + t*128;
      int r = e4 >> 3, c = (e4 & 7) * 4;
      int gr = m0 + r;
      f4 v = {0.f, 0.f, 0.f, 0.f};
      if (gr < M) v = *(const f4*)&A[(size_t)gr*K + kk + c];
      v4h hv = { (_Float16)v.x, (_Float16)v.y, (_Float16)v.z, (_Float16)v.w };
      *(v4h*)&As[r][c] = hv;
    }
  }
}
__device__ __forceinline__ v16h load_A_frag(const _Float16 (*As)[40], int wave, int lane){
  const int am = (lane & 15) + wave*16;
  const int kb = (lane >> 4) * 16;
  v8h lo = *(const v8h*)&As[am][kb];        // 16B aligned (row stride 80B)
  v8h hi = *(const v8h*)&As[am][kb + 8];
  return __builtin_shufflevector(lo, hi, 0,1,2,3,4,5,6,7,8,9,10,11,12,13,14,15);
}

// ---------------- GAT output GEMM + fused epilogue (head-avg + bias + residual + LN + relu) ----------------
__global__ void __launch_bounds__(128) k_gat_gemm(
    const float* __restrict__ z, const v16h* __restrict__ Bp,
    const float* __restrict__ bias, const float* __restrict__ hres,
    const float* __restrict__ gamma, const float* __restrict__ beta,
    float* __restrict__ hout, int M)
{
  __shared__ _Float16 As[64][40];
  __shared__ float    Cs[64][128];
  __shared__ float    mu[64], rs[64];

  const int tid  = threadIdx.x;
  const int wave = tid >> 5;
  const int lane = tid & 31;
  const int m0   = blockIdx.x * 64;

  v8f acc[8] = {};

  for (int kk = 0; kk < 1024; kk += 32){
    stage_A_tile(As, z, m0, M, 1024, kk, tid);
    __syncthreads();

    const int s = kk >> 5;
    v16h bf[8];
    #pragma unroll
    for (int t = 0; t < 8; ++t)                 // issue all B loads up front
      bf[t] = Bp[(size_t)(s*8 + t)*32 + lane];
    v16h af = load_A_frag(As, wave, lane);
    #pragma unroll
    for (int t = 0; t < 8; ++t)
      acc[t] = __builtin_amdgcn_wmma_f32_16x16x32_f16(false, af, false, bf[t],
                                                      (short)0, acc[t], false, false);
    __syncthreads();
  }

  #pragma unroll
  for (int t = 0; t < 8; ++t)
    #pragma unroll
    for (int r = 0; r < 8; ++r){
      int mr = wave*16 + r + 8*(lane >> 4);
      Cs[mr][t*16 + (lane & 15)] = acc[t][r];
    }
  __syncthreads();

  for (int r = 0; r < 64; ++r){
    int gr = m0 + r;
    if (gr < M) Cs[r][tid] = Cs[r][tid]*0.125f + bias[tid] + hres[(size_t)gr*HIDC + tid];
  }
  __syncthreads();
  if (tid < 64){
    int gr = m0 + tid;
    if (gr < M){
      float s = 0.f, s2 = 0.f;
      for (int c = 0; c < HIDC; ++c){ float v = Cs[tid][c]; s += v; s2 += v*v; }
      float m  = s * (1.0f/128.0f);
      float var = s2 * (1.0f/128.0f) - m*m;
      mu[tid] = m;
      rs[tid] = rsqrtf(var + 1e-5f);
    }
  }
  __syncthreads();
  for (int r = 0; r < 64; ++r){
    int gr = m0 + r;
    if (gr < M){
      float v = (Cs[r][tid] - mu[r]) * rs[r] * gamma[tid] + beta[tid];
      hout[(size_t)gr*HIDC + tid] = fmaxf(v, 0.f);
    }
  }
}

// ---------------- generic MLP GEMM: out = act(A[M,K] @ Wpacked[K,NTILE*16] + b) ----------------
template<int NTILE>
__global__ void __launch_bounds__(128) k_mlp_gemm(
    const float* __restrict__ A, const v16h* __restrict__ Bp,
    const float* __restrict__ bias, float* __restrict__ out,
    int M, int K, int relu)
{
  __shared__ _Float16 As[64][40];
  __shared__ float    Cs[64][128];

  const int tid  = threadIdx.x;
  const int wave = tid >> 5;
  const int lane = tid & 31;
  const int m0   = blockIdx.x * 64;
  const int Ncols = NTILE * 16;

  v8f acc[NTILE] = {};

  for (int kk = 0; kk < K; kk += 32){
    stage_A_tile(As, A, m0, M, K, kk, tid);
    __syncthreads();

    const int s = kk >> 5;
    v16h bf[NTILE];
    #pragma unroll
    for (int t = 0; t < NTILE; ++t)
      bf[t] = Bp[(size_t)(s*NTILE + t)*32 + lane];
    v16h af = load_A_frag(As, wave, lane);
    #pragma unroll
    for (int t = 0; t < NTILE; ++t)
      acc[t] = __builtin_amdgcn_wmma_f32_16x16x32_f16(false, af, false, bf[t],
                                                      (short)0, acc[t], false, false);
    __syncthreads();
  }

  #pragma unroll
  for (int t = 0; t < NTILE; ++t)
    #pragma unroll
    for (int r = 0; r < 8; ++r){
      int mr = wave*16 + r + 8*(lane >> 4);
      Cs[mr][t*16 + (lane & 15)] = acc[t][r];
    }
  __syncthreads();

  if (tid < Ncols){
    for (int r = 0; r < 64; ++r){
      int gr = m0 + r;
      if (gr < M){
        float v = Cs[r][tid] + bias[tid];
        if (relu) v = fmaxf(v, 0.f);
        out[(size_t)gr*Ncols + tid] = v;
      }
    }
  }
}

// ---------------- pooling + graph MLP ----------------
__global__ void k_pool_init(float* gsum, unsigned* gmax, float* gcnt){
  int i = threadIdx.x;
  if (i < 512){ gsum[i] = 0.f; gmax[i] = ENC_NEG_MAX; }
  if (i < 4) gcnt[i] = 0.f;
}
__global__ void k_pool_accum(const float* __restrict__ h, const int* __restrict__ batch,
                             float* __restrict__ gsum, unsigned* __restrict__ gmax,
                             float* __restrict__ gcnt, int N){
  int n = blockIdx.x, c = threadIdx.x;
  if (n >= N) return;
  int b = batch[n];
  float v = h[(size_t)n*HIDC + c];
  atomicAdd(&gsum[b*HIDC + c], v);
  atomicMax(&gmax[b*HIDC + c], fenc(v));
  if (c == 0) atomicAdd(&gcnt[b], 1.0f);
}
__global__ void k_graph_mlp(const float* __restrict__ gsum, const unsigned* __restrict__ gmax,
                            const float* __restrict__ gcnt,
                            const float* __restrict__ W1, const float* __restrict__ b1,
                            const float* __restrict__ W2, const float* __restrict__ b2,
                            float* __restrict__ out){
  __shared__ float g[4][256];
  __shared__ float t[4][128];
  int tid = threadIdx.x;
  for (int idx = tid; idx < 4*256; idx += 128){
    int b = idx >> 8, c = idx & 255;
    float cnt = gcnt[b];
    float v;
    if (c < 128) v = gsum[b*HIDC + c] / fmaxf(cnt, 1.0f);
    else         v = (cnt > 0.f) ? fdec(gmax[b*HIDC + (c - 128)]) : 0.f;
    g[b][c] = v;
  }
  __syncthreads();
  for (int idx = tid; idx < 4*128; idx += 128){
    int b = idx >> 7, c = idx & 127;
    float s = b1[c];
    for (int k = 0; k < 256; ++k) s += g[b][k] * W1[k*128 + c];
    t[b][c] = fmaxf(s, 0.f);
  }
  __syncthreads();
  for (int idx = tid; idx < 4*64; idx += 128){
    int b = idx >> 6, c = idx & 63;
    float s = b2[c];
    for (int k = 0; k < 128; ++k) s += t[b][k] * W2[k*64 + c];
    out[b*64 + c] = s;
  }
}

// ================= host orchestration =================
extern "C" void kernel_launch(void* const* d_in, const int* in_sizes, int n_in,
                              void* d_out, int out_size, void* d_ws, size_t ws_size,
                              hipStream_t stream) {
  const int N  = in_sizes[3];          // batch vector length = #nodes
  const int E  = in_sizes[1] / 4;      // D_EDGE = 4
  const int EN = E + N;

  const float* x         = (const float*)d_in[0];
  const float* edge_attr = (const float*)d_in[1];
  const int*   eidx      = (const int*)d_in[2];
  const int*   srcI      = eidx;
  const int*   dstI      = eidx + E;
  const int*   batch     = (const int*)d_in[3];
  const float* W_node = (const float*)d_in[4];
  const float* b_node = (const float*)d_in[5];
  const float* W_eenc = (const float*)d_in[6];
  const float* b_eenc = (const float*)d_in[7];
  const float* W_lin  = (const float*)d_in[8];
  const float* W_edge = (const float*)d_in[9];
  const float* att_src  = (const float*)d_in[10];
  const float* att_dst  = (const float*)d_in[11];
  const float* att_edge = (const float*)d_in[12];
  const float* bias_gat = (const float*)d_in[13];
  const float* ln_g = (const float*)d_in[14];
  const float* ln_b = (const float*)d_in[15];
  const float* W_no1 = (const float*)d_in[16];
  const float* b_no1 = (const float*)d_in[17];
  const float* W_no2 = (const float*)d_in[18];
  const float* b_no2 = (const float*)d_in[19];
  const float* W_go1 = (const float*)d_in[20];
  const float* b_go1 = (const float*)d_in[21];
  const float* W_go2 = (const float*)d_in[22];
  const float* b_go2 = (const float*)d_in[23];

  char* wsb = (char*)d_ws;
  size_t off = 0;
  auto carve = [&](size_t bytes) -> void* {
    void* p = wsb + off;
    off += (bytes + 255) & ~(size_t)255;
    return p;
  };

  float*     h0        = (float*)carve((size_t)N*HIDC*4);
  float*     h1        = (float*)carve((size_t)N*HIDC*4);
  float*     ea        = (float*)carve((size_t)E*HIDC*4);
  float*     loop_attr = (float*)carve((size_t)N*HIDC*4);
  float*     deg       = (float*)carve((size_t)N*4);
  float*     vsrc      = (float*)carve(1024*4);
  float*     vdst      = (float*)carve(1024*4);
  float*     vedge     = (float*)carve(1024*4);
  float*     asrc      = (float*)carve((size_t)N*8*4);
  float*     adst      = (float*)carve((size_t)N*8*4);
  float*     aed       = (float*)carve((size_t)EN*8*4);
  float*     exb       = (float*)carve((size_t)EN*8*4);
  unsigned*  maxb      = (unsigned*)carve((size_t)N*8*4);
  float*     sumb      = (float*)carve((size_t)N*8*4);
  float*     z         = (float*)carve((size_t)N*1024*4);
  _Float16*  Bcat      = (_Float16*)carve((size_t)1024*128*2);
  _Float16*  W1h       = (_Float16*)carve((size_t)128*128*2);
  _Float16*  W2h       = (_Float16*)carve((size_t)128*64*2);
  float*     t1        = (float*)carve((size_t)N*HIDC*4);
  float*     gsum      = (float*)carve(4*128*4);
  unsigned*  gmax      = (unsigned*)carve(4*128*4);
  float*     gcnt      = (float*)carve(4*4);

  // encoders
  k_encode<<<N, 128, 0, stream>>>(x, W_node, b_node, h0, N, 8);
  k_encode<<<E, 128, 0, stream>>>(edge_attr, W_eenc, b_eenc, ea, E, 4);

  // self-loop attr (layer-invariant)
  hipMemsetAsync(loop_attr, 0, (size_t)N*HIDC*4, stream);
  hipMemsetAsync(deg, 0, (size_t)N*4, stream);
  k_loop_accum<<<E, 128, 0, stream>>>(ea, dstI, loop_attr, deg, E);
  k_loop_div<<<N, 128, 0, stream>>>(loop_attr, deg, N);

  // pack fp16 weights for node MLP in WMMA fragment order
  k_pack_w<<<128, 128, 0, stream>>>(W_no1, W1h, 128);
  k_pack_w<<<128, 64,  0, stream>>>(W_no2, W2h, 64);

  float* hcur = h0;
  float* hnxt = h1;
  const int gElems = EN * 8, nElems = N * 8;
  for (int l = 0; l < 4; ++l){
    const float* Wl = W_lin  + (size_t)l*128*1024;
    const float* We = W_edge + (size_t)l*128*1024;
    k_attvec<<<8, 128, 0, stream>>>(Wl, att_src  + (size_t)l*1024, vsrc);
    k_attvec<<<8, 128, 0, stream>>>(Wl, att_dst  + (size_t)l*1024, vdst);
    k_attvec<<<8, 128, 0, stream>>>(We, att_edge + (size_t)l*1024, vedge);
    k_bcat_pack<<<1024, 128, 0, stream>>>(Wl, Bcat);

    k_node_logits<<<(nElems + 255)/256, 256, 0, stream>>>(hcur, vsrc, vdst, asrc, adst, N);
    k_edge_logits<<<(gElems + 255)/256, 256, 0, stream>>>(ea, loop_attr, vedge, aed, E, N);

    k_smax_init<<<(nElems + 255)/256, 256, 0, stream>>>(maxb, sumb, nElems);
    k_logit_max<<<(gElems + 255)/256, 256, 0, stream>>>(srcI, dstI, asrc, adst, aed, maxb, E, N);
    k_exp_sum<<<(gElems + 255)/256, 256, 0, stream>>>(dstI, aed, maxb, exb, sumb, E, N);

    hipMemsetAsync(z, 0, (size_t)N*1024*4, stream);
    k_aggregate<<<EN, 128, 0, stream>>>(srcI, dstI, hcur, exb, sumb, z, E, N);

    k_gat_gemm<<<(N + 63)/64, 128, 0, stream>>>(z, (const v16h*)Bcat,
                                                bias_gat + (size_t)l*128, hcur,
                                                ln_g + (size_t)l*128, ln_b + (size_t)l*128,
                                                hnxt, N);
    float* tmp = hcur; hcur = hnxt; hnxt = tmp;
  }

  // node MLP -> d_out[0 : N*64)
  k_mlp_gemm<8><<<(N + 63)/64, 128, 0, stream>>>(hcur, (const v16h*)W1h, b_no1, t1, N, 128, 1);
  k_mlp_gemm<4><<<(N + 63)/64, 128, 0, stream>>>(t1, (const v16h*)W2h, b_no2, (float*)d_out, N, 128, 0);

  // graph pooling + MLP -> d_out[N*64 : N*64 + 4*64)
  k_pool_init<<<1, 512, 0, stream>>>(gsum, gmax, gcnt);
  k_pool_accum<<<N, 128, 0, stream>>>(hcur, batch, gsum, gmax, gcnt, N);
  k_graph_mlp<<<1, 128, 0, stream>>>(gsum, gmax, gcnt, W_go1, b_go1, W_go2, b_go2,
                                     (float*)d_out + (size_t)N*64);
}
